// SpatialSelfAttention_47184510714015
// MI455X (gfx1250) — matrix operand order repaired and checked by reading, not compiled, not run on hardware
//
#include <hip/hip_runtime.h>
#include <hip/hip_bf16.h>
#include <math.h>

typedef __attribute__((ext_vector_type(16))) _Float16 v16h;
typedef __attribute__((ext_vector_type(8)))  float    v8f;

#define TPB 256
#define BK  32
#define LDST 72   // LDS row stride in halves; 144B rows (16B aligned)

// ---------------------------------------------------------------------------
// f32 -> f16 conversion (weights)
// ---------------------------------------------------------------------------
__global__ __launch_bounds__(TPB) void f32_to_f16_kernel(const float* __restrict__ in,
                                                         _Float16* __restrict__ out, int n) {
  int i = blockIdx.x * TPB + threadIdx.x;
  if (i < n) out[i] = (_Float16)in[i];
}

// ---------------------------------------------------------------------------
// GroupNorm: one block per (batch, group). group = 16 channels x P positions.
// Writes normalized activations as f16 for the WMMA GEMMs.
// ---------------------------------------------------------------------------
__global__ __launch_bounds__(TPB) void groupnorm_kernel(const float* __restrict__ x,
                                                        const float* __restrict__ gamma,
                                                        const float* __restrict__ beta,
                                                        _Float16* __restrict__ h,
                                                        int C, int P, int chPerGroup) {
  const int g = blockIdx.x & 31;          // NUM_GROUPS = 32
  const int b = blockIdx.x >> 5;
  const long base = ((long)b * C + (long)g * chPerGroup) * P;
  const int n = chPerGroup * P;
  const int tid = threadIdx.x;

  float s = 0.f, ss = 0.f;
  for (int i = tid; i < n; i += TPB) {
    float v = x[base + i];
    s += v; ss += v * v;
  }
  __shared__ float rs[TPB];
  __shared__ float rq[TPB];
  rs[tid] = s; rq[tid] = ss;
  __syncthreads();
  for (int st = TPB / 2; st > 0; st >>= 1) {
    if (tid < st) { rs[tid] += rs[tid + st]; rq[tid] += rq[tid + st]; }
    __syncthreads();
  }
  const float inv_n = 1.0f / (float)n;
  const float mean = rs[0] * inv_n;
  const float var  = rq[0] * inv_n - mean * mean;
  const float rstd = rsqrtf(var + 1e-6f);

  for (int i = tid; i < n; i += TPB) {
    int c = g * chPerGroup + i / P;
    float v = (x[base + i] - mean) * rstd * gamma[c] + beta[c];
    h[base + i] = (_Float16)v;
  }
}

// ---------------------------------------------------------------------------
// Generic tiled WMMA GEMM:  D = alpha * (A x B + bias) [+ residual]
//   A: f16, [M][K] row-major (TA=0) or [K][M] (TA=1)
//   B: f16, [K][N] row-major
//   D: f16 (OUTF16=1) or f32 with optional residual (OUTF16=0)
// Block: 256 threads = 8 waves; tile 64(M) x 64(N); waves 4x2, each 16x32.
// M, N, K assumed multiples of 64/64/32 (true here: 512, 4096).
// ---------------------------------------------------------------------------
template <int TA, int OUTF16>
__global__ __launch_bounds__(TPB)
void gemm_wmma(const _Float16* __restrict__ A, const _Float16* __restrict__ B,
               void* __restrict__ Dvp, const float* __restrict__ bias,
               const float* __restrict__ residual,
               int M, int N, int K,
               long sA, long sB, long sD, long sR, float alpha) {
  __shared__ _Float16 As[BK][LDST];   // As[k][m]
  __shared__ _Float16 Bs[BK][LDST];   // Bs[k][n]

  const int tid  = threadIdx.x;
  const int lane = tid & 31;
  const int l15  = lane & 15;
  const int hi   = lane >> 4;
  const int wave = tid >> 5;
  const int wm   = wave >> 1;   // 0..3 -> 16-row (M) tile
  const int wn   = wave & 1;    // 0..1 -> 32-col (N) tile
  const int bm   = blockIdx.y * 64;
  const int bn   = blockIdx.x * 64;
  const int bz   = blockIdx.z;

  const _Float16* Ab = A + (long)bz * sA;
  const _Float16* Bb = B + (long)bz * sB;

  v8f acc0 = {}; v8f acc1 = {};

  for (int k0 = 0; k0 < K; k0 += BK) {
    if (TA == 0) {
      const int m  = tid >> 2;            // 0..63
      const int kq = (tid & 3) << 3;      // 0,8,16,24
      _Float16 tmp[8];
      *(int4*)tmp = *(const int4*)(Ab + (long)(bm + m) * K + (k0 + kq));
#pragma unroll
      for (int e = 0; e < 8; ++e) As[kq + e][m] = tmp[e];
    } else {
      const int kk = tid >> 3;            // 0..31
      const int mq = (tid & 7) << 3;      // 0..56
      *(int4*)&As[kk][mq] = *(const int4*)(Ab + (long)(k0 + kk) * M + (bm + mq));
    }
    {
      const int kk = tid >> 3;
      const int nq = (tid & 7) << 3;
      *(int4*)&Bs[kk][nq] = *(const int4*)(Bb + (long)(k0 + kk) * N + (bn + nq));
    }
    __syncthreads();

    // A fragment: lanes 0-15 hold K=0..15, lanes 16-31 hold K=16..31
    v16h af;
#pragma unroll
    for (int r = 0; r < 4; ++r) {
      const int k = 16 * hi + 2 * r;
      af[2 * r]     = As[k][wm * 16 + l15];
      af[2 * r + 1] = As[k + 1][wm * 16 + l15];
    }
#pragma unroll
    for (int r = 4; r < 8; ++r) {
      const int k = 16 * hi + 2 * r;
      af[2 * r]     = As[k][wm * 16 + l15];
      af[2 * r + 1] = As[k + 1][wm * 16 + l15];
    }
    // B fragments: lanes 0-15 hold K=0..15, lanes 16-31 hold K=16..31
    v16h bf0, bf1;
#pragma unroll
    for (int e = 0; e < 16; ++e) {
      const int k = 16 * hi + e;
      bf0[e] = Bs[k][wn * 32 + l15];
      bf1[e] = Bs[k][wn * 32 + 16 + l15];
    }

    acc0 = __builtin_amdgcn_wmma_f32_16x16x32_f16(false, af, false, bf0, (short)0, acc0, false, false);
    acc1 = __builtin_amdgcn_wmma_f32_16x16x32_f16(false, af, false, bf1, (short)0, acc1, false, false);
    __syncthreads();
  }

  // C/D layout: VGPR r holds row m = r + 8*hi, column n = l15 (+tile offsets)
  const int gn0 = bn + wn * 32 + l15;
  const int gn1 = gn0 + 16;
#pragma unroll
  for (int r = 0; r < 8; ++r) {
    const int gm = bm + wm * 16 + 8 * hi + r;
    float v0 = acc0[r], v1 = acc1[r];
    if (bias) { const float bb = bias[gm]; v0 += bb; v1 += bb; }
    v0 *= alpha; v1 *= alpha;
    const long o0 = (long)gm * N + gn0;
    const long o1 = (long)gm * N + gn1;
    if (OUTF16) {
      _Float16* D = (_Float16*)Dvp + (long)bz * sD;
      D[o0] = (_Float16)v0;
      D[o1] = (_Float16)v1;
    } else {
      float* D = (float*)Dvp + (long)bz * sD;
      if (residual) {
        v0 += residual[(long)bz * sR + o0];
        v1 += residual[(long)bz * sR + o1];
      }
      D[o0] = v0;
      D[o1] = v1;
    }
  }
}

// ---------------------------------------------------------------------------
// Softmax row statistics over S[b][i][:]: m = max_j, l = sum_j exp(s - m)
// One block per (b,i) row.
// ---------------------------------------------------------------------------
__global__ __launch_bounds__(TPB) void rowstats_kernel(const float* __restrict__ S,
                                                       float* __restrict__ mrow,
                                                       float* __restrict__ lrow, int P) {
  const long row = blockIdx.x;
  const float* r = S + row * (long)P;
  const int tid = threadIdx.x;
  __shared__ float red[TPB];

  float m = -3.0e38f;
  for (int j = tid; j < P; j += TPB) m = fmaxf(m, r[j]);
  red[tid] = m;
  __syncthreads();
  for (int st = TPB / 2; st > 0; st >>= 1) {
    if (tid < st) red[tid] = fmaxf(red[tid], red[tid + st]);
    __syncthreads();
  }
  m = red[0];
  __syncthreads();

  float l = 0.f;
  for (int j = tid; j < P; j += TPB) l += __expf(r[j] - m);
  red[tid] = l;
  __syncthreads();
  for (int st = TPB / 2; st > 0; st >>= 1) {
    if (tid < st) red[tid] += red[tid + st];
    __syncthreads();
  }
  if (tid == 0) { mrow[row] = m; lrow[row] = red[0]; }
}

// ---------------------------------------------------------------------------
// O[b][c][i] = (1/l_i) * sum_j exp(S[b][i][j] - m_i) * V[b][c][j]
// WMMA GEMM with M=c, N=i, K=j; exp() applied during B-tile staging into LDS.
// ---------------------------------------------------------------------------
__global__ __launch_bounds__(TPB)
void attn_pv_wmma(const _Float16* __restrict__ V, const float* __restrict__ S,
                  const float* __restrict__ mrow, const float* __restrict__ lrow,
                  _Float16* __restrict__ O, int C, int P) {
  __shared__ _Float16 As[BK][LDST];   // As[j][c] from V
  __shared__ _Float16 Bs[BK][LDST];   // Bs[j][i] = exp(S[i][j]-m_i)

  const int tid  = threadIdx.x;
  const int lane = tid & 31;
  const int l15  = lane & 15;
  const int hi   = lane >> 4;
  const int wave = tid >> 5;
  const int wm   = wave >> 1;
  const int wn   = wave & 1;
  const int bm   = blockIdx.y * 64;   // channel tile
  const int bn   = blockIdx.x * 64;   // query (i) tile
  const int bz   = blockIdx.z;

  const _Float16* Vb = V + (long)bz * C * P;
  const float*    Sb = S + (long)bz * P * P;

  const int iLoc = tid >> 2;            // 0..63 (fixed per thread)
  const int jq   = (tid & 3) << 3;      // 0,8,16,24
  const float mi = mrow[(long)bz * P + bn + iLoc];

  v8f acc0 = {}; v8f acc1 = {};

  for (int k0 = 0; k0 < P; k0 += BK) {
    {   // stage V tile (64 c x 32 j), transpose into As[j][c]
      _Float16 tmp[8];
      *(int4*)tmp = *(const int4*)(Vb + (long)(bm + iLoc) * P + (k0 + jq));
#pragma unroll
      for (int e = 0; e < 8; ++e) As[jq + e][iLoc] = tmp[e];
    }
    {   // stage exp(S - m) tile into Bs[j][i]
      const float* sp = Sb + (long)(bn + iLoc) * P + (k0 + jq);
      float4 s0 = *(const float4*)(sp);
      float4 s1 = *(const float4*)(sp + 4);
      Bs[jq + 0][iLoc] = (_Float16)__expf(s0.x - mi);
      Bs[jq + 1][iLoc] = (_Float16)__expf(s0.y - mi);
      Bs[jq + 2][iLoc] = (_Float16)__expf(s0.z - mi);
      Bs[jq + 3][iLoc] = (_Float16)__expf(s0.w - mi);
      Bs[jq + 4][iLoc] = (_Float16)__expf(s1.x - mi);
      Bs[jq + 5][iLoc] = (_Float16)__expf(s1.y - mi);
      Bs[jq + 6][iLoc] = (_Float16)__expf(s1.z - mi);
      Bs[jq + 7][iLoc] = (_Float16)__expf(s1.w - mi);
    }
    __syncthreads();

    v16h af;
#pragma unroll
    for (int r = 0; r < 4; ++r) {
      const int k = 16 * hi + 2 * r;
      af[2 * r]     = As[k][wm * 16 + l15];
      af[2 * r + 1] = As[k + 1][wm * 16 + l15];
    }
#pragma unroll
    for (int r = 4; r < 8; ++r) {
      const int k = 16 * hi + 2 * r;
      af[2 * r]     = As[k][wm * 16 + l15];
      af[2 * r + 1] = As[k + 1][wm * 16 + l15];
    }
    v16h bf0, bf1;
#pragma unroll
    for (int e = 0; e < 16; ++e) {
      const int k = 16 * hi + e;
      bf0[e] = Bs[k][wn * 32 + l15];
      bf1[e] = Bs[k][wn * 32 + 16 + l15];
    }

    acc0 = __builtin_amdgcn_wmma_f32_16x16x32_f16(false, af, false, bf0, (short)0, acc0, false, false);
    acc1 = __builtin_amdgcn_wmma_f32_16x16x32_f16(false, af, false, bf1, (short)0, acc1, false, false);
    __syncthreads();
  }

  const int gn0 = bn + wn * 32 + l15;
  const int gn1 = gn0 + 16;
  const float r0 = 1.0f / lrow[(long)bz * P + gn0];
  const float r1 = 1.0f / lrow[(long)bz * P + gn1];
#pragma unroll
  for (int r = 0; r < 8; ++r) {
    const int gm = bm + wm * 16 + 8 * hi + r;
    O[((long)bz * C + gm) * P + gn0] = (_Float16)(acc0[r] * r0);
    O[((long)bz * C + gm) * P + gn1] = (_Float16)(acc1[r] * r1);
  }
}

// ---------------------------------------------------------------------------
// Host-side orchestration
// ---------------------------------------------------------------------------
extern "C" void kernel_launch(void* const* d_in, const int* in_sizes, int n_in,
                              void* d_out, int out_size, void* d_ws, size_t ws_size,
                              hipStream_t stream) {
  const float* x     = (const float*)d_in[0];
  const float* gamma = (const float*)d_in[1];
  const float* beta  = (const float*)d_in[2];
  const float* wq    = (const float*)d_in[3];
  const float* bq    = (const float*)d_in[4];
  const float* wk    = (const float*)d_in[5];
  const float* bk    = (const float*)d_in[6];
  const float* wv    = (const float*)d_in[7];
  const float* bv    = (const float*)d_in[8];
  const float* wo    = (const float*)d_in[9];
  const float* bo    = (const float*)d_in[10];

  const int B = 4;
  const int C = in_sizes[1];                 // 512
  const int P = in_sizes[0] / (B * C);       // 4096
  const long CP = (long)C * P;
  const long PP = (long)P * P;

  // workspace carve
  char* ws = (char*)d_ws;
  size_t off = 0;
  auto carve = [&](size_t bytes) {
    size_t o = off;
    off = (off + bytes + 255) & ~(size_t)255;
    return o;
  };
  _Float16* h16  = (_Float16*)(ws + carve((size_t)B * CP * 2));
  _Float16* q16  = (_Float16*)(ws + carve((size_t)B * CP * 2));
  _Float16* k16  = (_Float16*)(ws + carve((size_t)B * CP * 2));
  _Float16* v16  = (_Float16*)(ws + carve((size_t)B * CP * 2));
  _Float16* o16  = (_Float16*)(ws + carve((size_t)B * CP * 2));
  _Float16* wq16 = (_Float16*)(ws + carve((size_t)C * C * 2));
  _Float16* wk16 = (_Float16*)(ws + carve((size_t)C * C * 2));
  _Float16* wv16 = (_Float16*)(ws + carve((size_t)C * C * 2));
  _Float16* wo16 = (_Float16*)(ws + carve((size_t)C * C * 2));
  float*    Sbuf = (float*)(ws + carve((size_t)B * PP * 4));
  float*    mrow = (float*)(ws + carve((size_t)B * P * 4));
  float*    lrow = (float*)(ws + carve((size_t)B * P * 4));

  // 1) weights -> f16
  {
    int n = C * C;
    int g = (n + TPB - 1) / TPB;
    f32_to_f16_kernel<<<g, TPB, 0, stream>>>(wq, wq16, n);
    f32_to_f16_kernel<<<g, TPB, 0, stream>>>(wk, wk16, n);
    f32_to_f16_kernel<<<g, TPB, 0, stream>>>(wv, wv16, n);
    f32_to_f16_kernel<<<g, TPB, 0, stream>>>(wo, wo16, n);
  }

  // 2) GroupNorm -> h16
  groupnorm_kernel<<<B * 32, TPB, 0, stream>>>(x, gamma, beta, h16, C, P, C / 32);

  // 3) q/k/v projections (q gets the 1/sqrt(C) scale folded in)
  {
    dim3 grid(P / 64, C / 64, B);
    const float scale = 1.0f / sqrtf((float)C);
    gemm_wmma<0, 1><<<grid, TPB, 0, stream>>>(wq16, h16, (void*)q16, bq, nullptr,
                                              C, P, C, 0, CP, CP, 0, scale);
    gemm_wmma<0, 1><<<grid, TPB, 0, stream>>>(wk16, h16, (void*)k16, bk, nullptr,
                                              C, P, C, 0, CP, CP, 0, 1.0f);
    gemm_wmma<0, 1><<<grid, TPB, 0, stream>>>(wv16, h16, (void*)v16, bv, nullptr,
                                              C, P, C, 0, CP, CP, 0, 1.0f);
  }

  // 4) S = q^T k  (A = q stored [C][P] -> TA=1; f32 output)
  {
    dim3 grid(P / 64, P / 64, B);
    gemm_wmma<1, 0><<<grid, TPB, 0, stream>>>(q16, k16, (void*)Sbuf, nullptr, nullptr,
                                              P, P, C, CP, CP, PP, 0, 1.0f);
  }

  // 5) softmax row statistics
  rowstats_kernel<<<B * P, TPB, 0, stream>>>(Sbuf, mrow, lrow, P);

  // 6) O = softmax(S) V^T  (exp applied on the fly, /l in epilogue)
  {
    dim3 grid(P / 64, C / 64, B);
    attn_pv_wmma<<<grid, TPB, 0, stream>>>(v16, Sbuf, mrow, lrow, o16, C, P);
  }

  // 7) out = x + wo * O + bo  (f32 output + residual)
  {
    dim3 grid(P / 64, C / 64, B);
    gemm_wmma<0, 0><<<grid, TPB, 0, stream>>>(wo16, o16, d_out, bo, x,
                                              C, P, C, 0, CP, CP, CP, 1.0f);
  }
}
